// TokenFeatureEnhancer_35304631173186
// MI455X (gfx1250) — matrix operand order repaired and checked by reading, not compiled, not run on hardware
//
#include <hip/hip_runtime.h>
#include <hip/hip_bf16.h>
#include <math.h>

#define DD     256
#define KSEL   8
#define BBANK  32
#define NCLS   71
#define BATCH  1024

typedef __attribute__((ext_vector_type(16))) _Float16 v16h;
typedef __attribute__((ext_vector_type(8)))  _Float16 v8h;
typedef __attribute__((ext_vector_type(8)))  float    v8f;

// ---------------- WMMA fragment helpers (CDNA5 16x16x32 f16) ----------------

// A-matrix 16x32 f16 (ISA 7.12.2): lane m<16 -> rows, K split:
//   elems 0..7  = K  (lane<16 ? 0..7  : 8..15)
//   elems 8..15 = K  (lane<16 ? 16..23: 24..31)
__device__ __forceinline__ v16h load_a_frag(const _Float16* base, int ldh) {
  const int lane = threadIdx.x & 31;
  const int m    = lane & 15;
  const int o    = (lane >> 4) << 3;          // 0 or 8
  const _Float16* p = base + m * ldh + o;
  v8h lo = *reinterpret_cast<const v8h*>(p);       // 16B
  v8h hi = *reinterpret_cast<const v8h*>(p + 16);  // 16B
  v16h f;
#pragma unroll
  for (int i = 0; i < 8; ++i) { f[i] = lo[i]; f[i + 8] = hi[i]; }
  return f;
}

// B-matrix 32x16 f16: lane n holds column n; elems 0..7 K=(lane<16?0..7:8..15), elems 8..15 K=(lane<16?16..23:24..31).
// B[k][n] = W[n][k] (nn.Linear y = x W^T) -> two 16B reads per lane, 32B apart, of row-major f16 W.
__device__ __forceinline__ v16h load_b_frag(const _Float16* W, int n0, int k0, int ldh) {
  const int lane = threadIdx.x & 31;
  const int n    = lane & 15;
  const int kh   = (lane >> 4) << 3;          // 0 or 8
  const _Float16* p = W + (size_t)(n0 + n) * ldh + (k0 + kh);
  v8h lo = *reinterpret_cast<const v8h*>(p);
  v8h hi = *reinterpret_cast<const v8h*>(p + 16);
  v16h f;
#pragma unroll
  for (int i = 0; i < 8; ++i) { f[i] = lo[i]; f[i + 8] = hi[i]; }
  return f;
}

__device__ __forceinline__ v8f wmma16(v16h a, v16h b, v8f c) {
  return __builtin_amdgcn_wmma_f32_16x16x32_f16(
      /*neg_a=*/false, a, /*neg_b=*/false, b,
      /*c_mod=*/(short)0, c, /*reuse_a=*/false, /*reuse_b=*/false);
}

// One wave: 16x64 tile of  X[16 x 256] @ W^T  (A fragments hoisted; 32 wmma).
__device__ __forceinline__ void gemm_16x64(const _Float16* Ash, int lda,
                                           const _Float16* Wg, int n_base,
                                           v8f acc[4]) {
  v16h a[8];
#pragma unroll
  for (int kt = 0; kt < 8; ++kt) a[kt] = load_a_frag(Ash + kt * 32, lda);
#pragma unroll
  for (int nt = 0; nt < 4; ++nt) {
    v8f c = {};
#pragma unroll
    for (int kt = 0; kt < 8; ++kt) {
      v16h b = load_b_frag(Wg, n_base + nt * 16, kt * 32, DD);
      c = wmma16(a[kt], b, c);
    }
    acc[nt] = c;
  }
}

// One wave: 32x64 tile (BOTH 16-row M-tiles) so every B fragment feeds two
// v_wmma -> halves L2 weight traffic of the dominant stage-2 kernel. 64 wmma.
__device__ __forceinline__ void gemm_dual_32x64(const _Float16* Ash, int lda,
                                                const _Float16* Wg, int n_base,
                                                v8f acc0[4], v8f acc1[4]) {
#pragma unroll
  for (int nt = 0; nt < 4; ++nt) { acc0[nt] = (v8f){}; acc1[nt] = (v8f){}; }
#pragma unroll
  for (int kt = 0; kt < 8; ++kt) {
    v16h a0 = load_a_frag(Ash + kt * 32, lda);
    v16h a1 = load_a_frag(Ash + 16 * lda + kt * 32, lda);
#pragma unroll
    for (int nt = 0; nt < 4; ++nt) {
      v16h b = load_b_frag(Wg, n_base + nt * 16, kt * 32, DD);
      acc0[nt] = wmma16(a0, b, acc0[nt]);
      acc1[nt] = wmma16(a1, b, acc1[nt]);
    }
  }
}

__device__ __forceinline__ float gelu_exact(float x) {
  return 0.5f * x * (1.0f + erff(x * 0.70710678118654752440f));
}

// tanh via native v_exp_f32; clamp keeps e^(2x) finite.
__device__ __forceinline__ float fast_tanh(float x) {
  x = fminf(fmaxf(x, -15.f), 15.f);
  const float e = __expf(2.0f * x);
  return (e - 1.0f) / (e + 1.0f);
}

// ---------------- Precompute kernels ----------------

__global__ void mean_kernel(const float* __restrict__ bank,
                            float* __restrict__ mean_out) {
  const int c = blockIdx.x, d = threadIdx.x;
  float s = 0.f;
  for (int j = 0; j < BBANK; ++j) s += bank[((size_t)c * BBANK + j) * DD + d];
  mean_out[c * DD + d] = s * (1.0f / BBANK);
}

__global__ void cvt_kernel(const float* __restrict__ s,
                           _Float16* __restrict__ d, int n) {
  const int i = blockIdx.x * blockDim.x + threadIdx.x;
  if (i < n) d[i] = (_Float16)s[i];
}

// One block (64 threads) per batch row: d^2 to all 71 class means, then top-8
// smallest (ties -> lowest index, matching top_k on -dist; sqrt is monotone).
__global__ __launch_bounds__(64) void topk_kernel(const float* __restrict__ target,
                                                  const float* __restrict__ mean,
                                                  int* __restrict__ idx_out) {
  __shared__ float xt[DD];
  __shared__ float d2[NCLS];
  const int b = blockIdx.x, t = threadIdx.x;
  for (int d = t; d < DD; d += 64) xt[d] = target[(size_t)b * DD + d];
  __syncthreads();
  for (int c = t; c < NCLS; c += 64) {
    float acc = 0.f;
    const float* m = mean + c * DD;
    for (int d = 0; d < DD; ++d) { float df = xt[d] - m[d]; acc += df * df; }
    d2[c] = acc;
  }
  __syncthreads();
  if (t == 0) {
    bool used[NCLS];
    for (int c = 0; c < NCLS; ++c) used[c] = false;
    for (int k = 0; k < KSEL; ++k) {
      int best = 0; float bv = 3.4e38f;
      for (int c = 0; c < NCLS; ++c)
        if (!used[c] && d2[c] < bv) { bv = d2[c]; best = c; }
      used[best] = true;
      idx_out[b * KSEL + k] = best;
    }
  }
}

// ---------------- Stage 1: 4 batch items x K=8 rows per block ----------------

__global__ __launch_bounds__(256, 1) void stage1_kernel(
    const float* __restrict__ target, const float* __restrict__ mean,
    const int* __restrict__ idxb,
    const _Float16* __restrict__ We1, const float* __restrict__ be1,
    const _Float16* __restrict__ Wf1,
    const _Float16* __restrict__ Wo1, const float* __restrict__ bo1,
    float* __restrict__ s1_out) {
  __shared__ alignas(16) union SU { _Float16 x[32][264]; float w[32][260]; } U;
  __shared__ alignas(16) _Float16 sh_e[32][264];
  __shared__ float sh_t[4][DD];

  const int tid  = threadIdx.x;
  const int lane = tid & 31;
  const int wid  = tid >> 5;
  const int wm   = wid & 1;        // 2 M-tiles of 16 rows
  const int wn   = wid >> 1;       // 4 column groups of 64
  const int hi8  = (lane >> 4) << 3;
  const int cn   = lane & 15;
  const int b0   = blockIdx.x * 4;

  // Phase A: A = fea_mean[idx] - target ; keep target rows for epilogues.
  for (int bl = 0; bl < 4; ++bl)
    sh_t[bl][tid] = target[(size_t)(b0 + bl) * DD + tid];
  for (int j = 0; j < 32; ++j) {
    const int bl = j >> 3;
    const int c  = idxb[(b0 + bl) * KSEL + (j & 7)];
    U.x[j][tid] = (_Float16)(mean[(size_t)c * DD + tid] - sh_t[bl][tid]);
  }
  __syncthreads();

  // GEMM1: e1 = gelu(x @ We1^T + be1)
  {
    v8f acc[4];
    gemm_16x64(&U.x[wm * 16][0], 264, We1, wn * 64, acc);
#pragma unroll
    for (int nt = 0; nt < 4; ++nt) {
      const int col = wn * 64 + nt * 16 + cn;
      const float bb = be1[col];
#pragma unroll
      for (int i = 0; i < 8; ++i)
        sh_e[wm * 16 + hi8 + i][col] = (_Float16)gelu_exact(acc[nt][i] + bb);
    }
  }
  __syncthreads();

  // GEMM2: w1 = e1 @ Wf1^T  (f32 logits into union region; U.x is dead)
  {
    v8f acc[4];
    gemm_16x64(&sh_e[wm * 16][0], 264, Wf1, wn * 64, acc);
#pragma unroll
    for (int nt = 0; nt < 4; ++nt) {
      const int col = wn * 64 + nt * 16 + cn;
#pragma unroll
      for (int i = 0; i < 8; ++i)
        U.w[wm * 16 + hi8 + i][col] = acc[nt][i];
    }
  }
  __syncthreads();

  // Softmax over K=8 rows per (batch item, column); scale e1 in place.
  for (int bl = 0; bl < 4; ++bl) {
    float v[8], mx = -3.4e38f;
#pragma unroll
    for (int k = 0; k < 8; ++k) { v[k] = U.w[bl * 8 + k][tid]; mx = fmaxf(mx, v[k]); }
    float s = 0.f;
#pragma unroll
    for (int k = 0; k < 8; ++k) { v[k] = __expf(v[k] - mx); s += v[k]; }
    const float inv = 1.0f / s;
#pragma unroll
    for (int k = 0; k < 8; ++k) {
      const int r = bl * 8 + k;
      sh_e[r][tid] = (_Float16)((float)sh_e[r][tid] * v[k] * inv);
    }
  }
  __syncthreads();

  // GEMM3 input: x3 = target + e1 (logits fully consumed -> reuse union as f16)
  for (int j = 0; j < 32; ++j)
    U.x[j][tid] = (_Float16)(sh_t[j >> 3][tid] + (float)sh_e[j][tid]);
  __syncthreads();

  // GEMM3: s1 = (1 + tanh(x3 @ Wo1^T + bo1)) * target
  {
    v8f acc[4];
    gemm_16x64(&U.x[wm * 16][0], 264, Wo1, wn * 64, acc);
#pragma unroll
    for (int nt = 0; nt < 4; ++nt) {
      const int col = wn * 64 + nt * 16 + cn;
      const float bb = bo1[col];
#pragma unroll
      for (int i = 0; i < 8; ++i) {
        const int row = wm * 16 + hi8 + i;
        const float off = fast_tanh(acc[nt][i] + bb);
        s1_out[((size_t)blockIdx.x * 32 + row) * DD + col] =
            (1.0f + off) * sh_t[row >> 3][col];
      }
    }
  }
}

// ------- Stage 2: one (b,k) pair per block; 4 waves, dual-M B-frag reuse -------

__global__ __launch_bounds__(128, 1) void stage2_kernel(
    const float* __restrict__ bank, const int* __restrict__ idxb,
    const float* __restrict__ s1,
    const _Float16* __restrict__ We2, const float* __restrict__ be2,
    const _Float16* __restrict__ Wf2,
    const _Float16* __restrict__ Wo2, const float* __restrict__ bo2,
    float* __restrict__ s2_out) {
  __shared__ alignas(16) union SU2 { _Float16 x[32][264]; float w[32][260]; } U;
  __shared__ alignas(16) _Float16 sh_e[32][264];
  __shared__ float sh_s1[DD];
  __shared__ float sh_red[2][DD];

  const int tid  = threadIdx.x;              // 0..127, 4 waves
  const int lane = tid & 31;
  const int wn   = tid >> 5;                 // 4 column groups of 64
  const int hi8  = (lane >> 4) << 3;
  const int cn   = lane & 15;
  const int bk   = blockIdx.x;               // b*K + k
  const int cls  = idxb[bk];

  for (int d = tid; d < DD; d += 128) sh_s1[d] = s1[(size_t)bk * DD + d];
  __syncthreads();
  for (int j = 0; j < 32; ++j)
    for (int d = tid; d < DD; d += 128)
      U.x[j][d] = (_Float16)(bank[((size_t)cls * BBANK + j) * DD + d] - sh_s1[d]);
  __syncthreads();

  // GEMM1: e2 = gelu(x2 @ We2^T + be2)
  {
    v8f a0[4], a1[4];
    gemm_dual_32x64(&U.x[0][0], 264, We2, wn * 64, a0, a1);
#pragma unroll
    for (int nt = 0; nt < 4; ++nt) {
      const int col = wn * 64 + nt * 16 + cn;
      const float bb = be2[col];
#pragma unroll
      for (int i = 0; i < 8; ++i) {
        sh_e[hi8 + i][col]      = (_Float16)gelu_exact(a0[nt][i] + bb);
        sh_e[16 + hi8 + i][col] = (_Float16)gelu_exact(a1[nt][i] + bb);
      }
    }
  }
  __syncthreads();

  // GEMM2: w2 = e2 @ Wf2^T
  {
    v8f a0[4], a1[4];
    gemm_dual_32x64(&sh_e[0][0], 264, Wf2, wn * 64, a0, a1);
#pragma unroll
    for (int nt = 0; nt < 4; ++nt) {
      const int col = wn * 64 + nt * 16 + cn;
#pragma unroll
      for (int i = 0; i < 8; ++i) {
        U.w[hi8 + i][col]      = a0[nt][i];
        U.w[16 + hi8 + i][col] = a1[nt][i];
      }
    }
  }
  __syncthreads();

  // Softmax over BANK=32 rows per column; scale e2.
  for (int d = tid; d < DD; d += 128) {
    float mx = -3.4e38f;
    for (int j = 0; j < 32; ++j) mx = fmaxf(mx, U.w[j][d]);
    float s = 0.f;
    for (int j = 0; j < 32; ++j) s += __expf(U.w[j][d] - mx);
    const float inv = 1.0f / s;
    for (int j = 0; j < 32; ++j)
      sh_e[j][d] = (_Float16)((float)sh_e[j][d] * __expf(U.w[j][d] - mx) * inv);
  }
  __syncthreads();
  for (int j = 0; j < 32; ++j)
    for (int d = tid; d < DD; d += 128)
      U.x[j][d] = (_Float16)(sh_s1[d] + (float)sh_e[j][d]);
  __syncthreads();

  // GEMM3: off2 = tanh(x @ Wo2^T + bo2); s2 = sum_j (1+off2)*s1
  {
    v8f a0[4], a1[4];
    gemm_dual_32x64(&U.x[0][0], 264, Wo2, wn * 64, a0, a1);
    const int m2 = lane >> 4;
#pragma unroll
    for (int nt = 0; nt < 4; ++nt) {
      const int col = wn * 64 + nt * 16 + cn;
      const float bb = bo2[col];
      float ps = 0.f;
#pragma unroll
      for (int i = 0; i < 8; ++i) {
        ps += 1.0f + fast_tanh(a0[nt][i] + bb);
        ps += 1.0f + fast_tanh(a1[nt][i] + bb);
      }
      sh_red[m2][col] = ps * sh_s1[col];   // (m2,col) unique per lane
    }
  }
  __syncthreads();
  for (int d = tid; d < DD; d += 128)
    s2_out[(size_t)bk * DD + d] = sh_red[0][d] + sh_red[1][d];
}

// ---------------- Host launcher ----------------

extern "C" void kernel_launch(void* const* d_in, const int* in_sizes, int n_in,
                              void* d_out, int out_size, void* d_ws, size_t ws_size,
                              hipStream_t stream) {
  (void)in_sizes; (void)n_in; (void)out_size; (void)ws_size;
  const float* target = (const float*)d_in[0];
  const float* bank   = (const float*)d_in[1];
  const float* W_e1   = (const float*)d_in[2];
  const float* b_e1   = (const float*)d_in[3];
  const float* W_o1   = (const float*)d_in[4];
  const float* b_o1   = (const float*)d_in[5];
  const float* W_e2   = (const float*)d_in[6];
  const float* b_e2   = (const float*)d_in[7];
  const float* W_o2   = (const float*)d_in[8];
  const float* b_o2   = (const float*)d_in[9];
  const float* W_f1   = (const float*)d_in[10];
  const float* W_f2   = (const float*)d_in[11];
  float* out = (float*)d_out;

  char* ws = (char*)d_ws;
  float*    fea_mean = (float*)ws;                       // 72,704 B
  int*      idxb     = (int*)(ws + 73728);               // 32,768 B
  _Float16* w16      = (_Float16*)(ws + 73728 + 32768);  // 6 x 131,072 B
  _Float16* We1h = w16 + 0 * 65536;
  _Float16* Wo1h = w16 + 1 * 65536;
  _Float16* We2h = w16 + 2 * 65536;
  _Float16* Wo2h = w16 + 3 * 65536;
  _Float16* Wf1h = w16 + 4 * 65536;
  _Float16* Wf2h = w16 + 5 * 65536;

  mean_kernel<<<NCLS, DD, 0, stream>>>(bank, fea_mean);
  cvt_kernel<<<256, 256, 0, stream>>>(W_e1, We1h, 65536);
  cvt_kernel<<<256, 256, 0, stream>>>(W_o1, Wo1h, 65536);
  cvt_kernel<<<256, 256, 0, stream>>>(W_e2, We2h, 65536);
  cvt_kernel<<<256, 256, 0, stream>>>(W_o2, Wo2h, 65536);
  cvt_kernel<<<256, 256, 0, stream>>>(W_f1, Wf1h, 65536);
  cvt_kernel<<<256, 256, 0, stream>>>(W_f2, Wf2h, 65536);
  topk_kernel<<<BATCH, 64, 0, stream>>>(target, fea_mean, idxb);
  stage1_kernel<<<BATCH / 4, 256, 0, stream>>>(target, fea_mean, idxb,
                                               We1h, b_e1, Wf1h, Wo1h, b_o1, out);
  stage2_kernel<<<BATCH * KSEL, 128, 0, stream>>>(bank, idxb, out,
                                                  We2h, b_e2, Wf2h, Wo2h, b_o2,
                                                  out + (size_t)BATCH * KSEL * DD);
}
